// CausalSelfAttention_20109036879984
// MI455X (gfx1250) — compile-verified, matrix operand order checked by reading, not run on hardware
//
#include <hip/hip_runtime.h>
#include <hip/hip_bf16.h>
#include <cstdint>

// ---------------------------------------------------------------------------
// Problem constants:  B=2, T=2048, C=2048, NH=16, NKV=8, HD=128.
// qkv width = (NH+2*NKV)*HD = 4096.
// ---------------------------------------------------------------------------
static constexpr int Bc   = 2;
static constexpr int Tc   = 2048;
static constexpr int Cc   = 2048;
static constexpr int NH   = 16;
static constexpr int NKV  = 8;
static constexpr int HD   = 128;
static constexpr int QKVW = (NH + 2 * NKV) * HD;   // 4096

typedef __attribute__((ext_vector_type(16))) __bf16 v16bf;
typedef __attribute__((ext_vector_type(8)))  float  v8f;

union FragB16 {
    v16bf bf;
    unsigned short us[16];
    uint4 q[2];
};

__device__ __forceinline__ unsigned short f2bf(float f) {
    unsigned int u = __float_as_uint(f);
    u += 0x7fffu + ((u >> 16) & 1u);       // round-to-nearest-even
    return (unsigned short)(u >> 16);
}

// ---------------------------------------------------------------------------
// One-shot cast kernels (run once; ~84MB total -> ~7us at 23.3 TB/s).
// ---------------------------------------------------------------------------
__global__ __launch_bounds__(256) void cast_bf16(
    const float* __restrict__ src, unsigned short* __restrict__ dst, int n)
{
    const int i = blockIdx.x * blockDim.x + threadIdx.x;
    if (i < n) dst[i] = f2bf(src[i]);
}

// src [rows, cols] fp32 -> dst [cols, rows] bf16 (coalesced read)
__global__ __launch_bounds__(256) void cast_transpose_bf16(
    const float* __restrict__ src, unsigned short* __restrict__ dst,
    int rows, int cols)
{
    const int i = blockIdx.x * blockDim.x + threadIdx.x;
    if (i >= rows * cols) return;
    const int r = i / cols, c = i - r * cols;
    dst[(size_t)c * rows + r] = f2bf(src[i]);
}

// ---------------------------------------------------------------------------
// bf16 WMMA GEMM:  C[M,N] = A[M,K] * Bt[N,K]^T  (A,Bt bf16, C fp32).
// Block tile 128x128, K-tile 32.  8 waves (4x2); each wave owns 32x64 =
// 2x4 WMMA 16x16 f32 accumulators.  Both LDS tiles are row-major [row][k]
// with stride 40 ushorts (20 dwords: bank-spread + 16B aligned), so every
// stage is a pair of b128 copies and every fragment is a pair of
// ds_load_b128.  Next K-tile is prefetched (global_prefetch_b8).
// ---------------------------------------------------------------------------
__global__ __launch_bounds__(256) void gemm_bf16_wmma(
    const unsigned short* __restrict__ A, const unsigned short* __restrict__ Bt,
    float* __restrict__ Cm, int M, int N, int K)
{
    __shared__ unsigned short As[128 * 40];   // [m-row][k]
    __shared__ unsigned short Bs[128 * 40];   // [n-row][k]

    const int tid  = threadIdx.x;
    const int lane = tid & 31;
    const int wave = tid >> 5;
    const int m0 = blockIdx.x * 128;
    const int n0 = blockIdx.y * 128;
    const int wr = wave >> 1;                 // 0..3 -> rows wr*32
    const int wc = wave & 1;                  // 0..1 -> cols wc*64
    const int hl = lane >> 4;
    const int lr = lane & 15;

    const int srow = tid >> 1;                // staging row 0..127
    const int skk  = (tid & 1) * 16;          // staging k-offset 0 / 16
    const unsigned short* pa = A  + (size_t)(m0 + srow) * K + skk;
    const unsigned short* pb = Bt + (size_t)(n0 + srow) * K + skk;

    v8f acc[2][4] = {};

    for (int k0 = 0; k0 < K; k0 += 32) {
        // --- stage tiles: pure b128 copies (no conversion, no transpose)
        *(uint4*)&As[srow * 40 + skk]     = *(const uint4*)(pa + k0);
        *(uint4*)&As[srow * 40 + skk + 8] = *(const uint4*)(pa + k0 + 8);
        *(uint4*)&Bs[srow * 40 + skk]     = *(const uint4*)(pb + k0);
        *(uint4*)&Bs[srow * 40 + skk + 8] = *(const uint4*)(pb + k0 + 8);
        if (k0 + 32 < K) {                     // prefetch next K-tile
            __builtin_prefetch(pa + k0 + 32, 0, 0);
            __builtin_prefetch(pb + k0 + 32, 0, 0);
        }
        __syncthreads();

        FragB16 af[2], bfm[4];
        #pragma unroll
        for (int mt = 0; mt < 2; ++mt) {
            const unsigned short* p = &As[(wr * 32 + mt * 16 + lr) * 40 + hl * 16];
            af[mt].q[0] = *(const uint4*)p;
            af[mt].q[1] = *(const uint4*)(p + 8);
        }
        #pragma unroll
        for (int nt = 0; nt < 4; ++nt) {
            const unsigned short* p = &Bs[(wc * 64 + nt * 16 + lr) * 40 + hl * 16];
            bfm[nt].q[0] = *(const uint4*)p;
            bfm[nt].q[1] = *(const uint4*)(p + 8);
        }
        #pragma unroll
        for (int mt = 0; mt < 2; ++mt)
            #pragma unroll
            for (int nt = 0; nt < 4; ++nt)
                acc[mt][nt] = __builtin_amdgcn_wmma_f32_16x16x32_bf16(
                    false, af[mt].bf, false, bfm[nt].bf,
                    (short)0, acc[mt][nt], false, false);
        __syncthreads();
    }

    #pragma unroll
    for (int mt = 0; mt < 2; ++mt)
        #pragma unroll
        for (int nt = 0; nt < 4; ++nt)
            #pragma unroll
            for (int r = 0; r < 8; ++r) {
                const int row = m0 + wr * 32 + mt * 16 + r + 8 * hl;
                const int col = n0 + wc * 64 + nt * 16 + lr;
                Cm[(size_t)row * N + col] = acc[mt][nt][r];
            }
}

// ---------------------------------------------------------------------------
// RoPE + split + head-major transpose + bf16 cast.
// qkv fp32 [B*T, 4096] -> Qb [B,NH,T,HD], Kb/Vb [B,NKV,T,HD] (bf16).
// ---------------------------------------------------------------------------
__global__ __launch_bounds__(256) void rope_split(
    const float* __restrict__ qkv,
    const float* __restrict__ cf, const float* __restrict__ sf,
    unsigned short* __restrict__ Qb, unsigned short* __restrict__ Kb,
    unsigned short* __restrict__ Vb)
{
    const int idx = blockIdx.x * blockDim.x + threadIdx.x;
    if (idx >= Bc * Tc * (QKVW / 2)) return;
    const int p    = idx & (QKVW / 2 - 1);
    const int row  = idx >> 11;               // b*T + t
    const int t    = row & (Tc - 1);
    const int b    = row >> 11;
    const int col0 = p * 2;

    const float x0 = qkv[(size_t)row * QKVW + col0];
    const float x1 = qkv[(size_t)row * QKVW + col0 + 1];

    if (col0 < NH * HD) {                      // Q + RoPE
        const int h  = col0 >> 7;
        const int hd = col0 & (HD - 1);
        const int j  = hd >> 1;
        const float c = cf[t * (HD / 2) + j], s = sf[t * (HD / 2) + j];
        unsigned short* dst = Qb + (((size_t)b * NH + h) * Tc + t) * HD + hd;
        dst[0] = f2bf(x0 * c - x1 * s);
        dst[1] = f2bf(x0 * s + x1 * c);
    } else if (col0 < (NH + NKV) * HD) {       // K + RoPE
        const int cc = col0 - NH * HD;
        const int h  = cc >> 7;
        const int hd = cc & (HD - 1);
        const int j  = hd >> 1;
        const float c = cf[t * (HD / 2) + j], s = sf[t * (HD / 2) + j];
        unsigned short* dst = Kb + (((size_t)b * NKV + h) * Tc + t) * HD + hd;
        dst[0] = f2bf(x0 * c - x1 * s);
        dst[1] = f2bf(x0 * s + x1 * c);
    } else {                                   // V, plain cast
        const int cc = col0 - (NH + NKV) * HD;
        const int h  = cc >> 7;
        const int hd = cc & (HD - 1);
        unsigned short* dst = Vb + (((size_t)b * NKV + h) * Tc + t) * HD + hd;
        dst[0] = f2bf(x0);
        dst[1] = f2bf(x1);
    }
}

// ---------------------------------------------------------------------------
// Flash attention with WMMA.  Grid (T/64, B*NH), 128 threads = 4 waves;
// each wave owns 16 query rows, Q in registers as 4 bf16 A-fragments.
// Per 32-key block: S = Q@K^T (8 WMMAs), streaming softmax (shfl_xor row
// reductions within 16-lane halves), P via LDS -> A-fragment, O += P@V
// (8 WMMAs).  Output written bf16 (feeds the projection GEMM directly).
// ---------------------------------------------------------------------------
__global__ __launch_bounds__(128) void attn_wmma(
    const unsigned short* __restrict__ Qb,
    const unsigned short* __restrict__ Kb,
    const unsigned short* __restrict__ Vb,
    unsigned short* __restrict__ O)
{
    __shared__ unsigned short Ks[32 * 136];    // [key][hd], stride 136
    __shared__ unsigned short Vs[128 * 40];    // [hd][key] transposed, stride 40
    __shared__ unsigned short Ps[4 * 16 * 40]; // per-wave P staging

    const int tid  = threadIdx.x;
    const int lane = tid & 31;
    const int w    = tid >> 5;
    const int hl   = lane >> 4;
    const int lr   = lane & 15;

    const int qblk = blockIdx.x * 64;
    const int bh   = blockIdx.y;
    const int b    = bh / NH;
    const int h    = bh % NH;
    const int g    = h / (NH / NKV);           // GQA kv head

    const unsigned short* Qp = Qb + (((size_t)b * NH  + h) * Tc + qblk + w * 16) * HD;
    const unsigned short* Kp = Kb + (((size_t)b * NKV + g) * Tc) * HD;
    const unsigned short* Vp = Vb + (((size_t)b * NKV + g) * Tc) * HD;

    FragB16 qf[4];
    #pragma unroll
    for (int c = 0; c < 4; ++c) {
        const unsigned short* p = Qp + (size_t)lr * HD + c * 32 + hl * 16;
        qf[c].q[0] = *(const uint4*)p;
        qf[c].q[1] = *(const uint4*)(p + 8);
    }

    v8f acc[8] = {};
    float mrow[8], lrow[8];
    #pragma unroll
    for (int r = 0; r < 8; ++r) { mrow[r] = -1e30f; lrow[r] = 0.0f; }

    const float scale = 0.08838834764831845f;  // 1/sqrt(128)
    const int kend = qblk + 64;

    for (int kb = 0; kb < kend; kb += 32) {
        for (int i = tid; i < 512; i += 128) { // K block, b128 copies
            const int key = i >> 4;
            const int hc  = (i & 15) * 8;
            *(uint4*)&Ks[key * 136 + hc] =
                *(const uint4*)&Kp[(size_t)(kb + key) * HD + hc];
        }
        for (int i = tid; i < 2048; i += 128) { // V block, transposed
            const int key = i >> 6;
            const int hp  = (i & 63) * 2;
            const unsigned int vp =
                *(const unsigned int*)&Vp[(size_t)(kb + key) * HD + hp];
            Vs[hp * 40 + key]       = (unsigned short)(vp & 0xffffu);
            Vs[(hp + 1) * 40 + key] = (unsigned short)(vp >> 16);
        }
        __syncthreads();

        v8f s[2] = {};
        #pragma unroll
        for (int nt = 0; nt < 2; ++nt)
            #pragma unroll
            for (int c = 0; c < 4; ++c) {
                FragB16 kf;
                const unsigned short* p = &Ks[(nt * 16 + lr) * 136 + c * 32 + hl * 16];
                kf.q[0] = *(const uint4*)p;
                kf.q[1] = *(const uint4*)(p + 8);
                s[nt] = __builtin_amdgcn_wmma_f32_16x16x32_bf16(
                    false, qf[c].bf, false, kf.bf, (short)0, s[nt], false, false);
            }

        float mnew[8];
        #pragma unroll
        for (int r = 0; r < 8; ++r) {
            const int qg = qblk + w * 16 + r + 8 * hl;
            float v0 = (kb + lr      <= qg) ? s[0][r] * scale : -1e30f;
            float v1 = (kb + 16 + lr <= qg) ? s[1][r] * scale : -1e30f;
            s[0][r] = v0; s[1][r] = v1;
            float m = fmaxf(v0, v1);
            #pragma unroll
            for (int off = 1; off < 16; off <<= 1)
                m = fmaxf(m, __shfl_xor(m, off, 32));
            mnew[r] = fmaxf(mrow[r], m);
        }
        #pragma unroll
        for (int r = 0; r < 8; ++r) {
            const float p0 = __expf(s[0][r] - mnew[r]);
            const float p1 = __expf(s[1][r] - mnew[r]);
            s[0][r] = p0; s[1][r] = p1;
            float sum = p0 + p1;
            #pragma unroll
            for (int off = 1; off < 16; off <<= 1)
                sum += __shfl_xor(sum, off, 32);
            const float alpha = __expf(mrow[r] - mnew[r]);
            lrow[r] = lrow[r] * alpha + sum;
            mrow[r] = mnew[r];
            #pragma unroll
            for (int nt = 0; nt < 8; ++nt) acc[nt][r] *= alpha;
        }

        #pragma unroll
        for (int nt = 0; nt < 2; ++nt)
            #pragma unroll
            for (int r = 0; r < 8; ++r)
                Ps[(w * 16 + r + 8 * hl) * 40 + nt * 16 + lr] = f2bf(s[nt][r]);
        asm volatile("s_wait_dscnt 0" ::: "memory"); // same-wave LDS RAW fence

        FragB16 pf;
        {
            const unsigned short* p = &Ps[(w * 16 + lr) * 40 + hl * 16];
            pf.q[0] = *(const uint4*)p;
            pf.q[1] = *(const uint4*)(p + 8);
        }

        #pragma unroll
        for (int nt = 0; nt < 8; ++nt) {
            FragB16 vf;
            const unsigned short* p = &Vs[(nt * 16 + lr) * 40 + hl * 16];
            vf.q[0] = *(const uint4*)p;
            vf.q[1] = *(const uint4*)(p + 8);
            acc[nt] = __builtin_amdgcn_wmma_f32_16x16x32_bf16(
                false, pf.bf, false, vf.bf, (short)0, acc[nt], false, false);
        }
        __syncthreads();
    }

    // epilogue: bf16 O[b, t, h*HD + hd] = acc / l   (feeds projection GEMM)
    #pragma unroll
    for (int nt = 0; nt < 8; ++nt)
        #pragma unroll
        for (int r = 0; r < 8; ++r) {
            const int qg  = qblk + w * 16 + r + 8 * hl;
            const int col = h * HD + nt * 16 + lr;
            O[((size_t)b * Tc + qg) * Cc + col] = f2bf(acc[nt][r] / lrow[r]);
        }
}

// ---------------------------------------------------------------------------
// Launcher.  Inputs: x, w_attn, w_proj, freqs_cos, freqs_sin (fp32).
// Workspace: qkv f32 | Qb | Kb | Vb | Obf | xb | waT | wpT  (bf16 = ushort).
// ---------------------------------------------------------------------------
extern "C" void kernel_launch(void* const* d_in, const int* in_sizes, int n_in,
                              void* d_out, int out_size, void* d_ws, size_t ws_size,
                              hipStream_t stream) {
    const float* x      = (const float*)d_in[0];
    const float* w_attn = (const float*)d_in[1];
    const float* w_proj = (const float*)d_in[2];
    const float* f_cos  = (const float*)d_in[3];
    const float* f_sin  = (const float*)d_in[4];
    float* out = (float*)d_out;

    const int M = Bc * Tc;                    // 4096

    char* ws = (char*)d_ws;
    float* qkv = (float*)ws;                   ws += (size_t)M * QKVW * sizeof(float);
    unsigned short* Qb  = (unsigned short*)ws; ws += (size_t)Bc * NH  * Tc * HD * 2;
    unsigned short* Kb  = (unsigned short*)ws; ws += (size_t)Bc * NKV * Tc * HD * 2;
    unsigned short* Vb  = (unsigned short*)ws; ws += (size_t)Bc * NKV * Tc * HD * 2;
    unsigned short* Obf = (unsigned short*)ws; ws += (size_t)M * Cc * 2;
    unsigned short* xb  = (unsigned short*)ws; ws += (size_t)M * Cc * 2;
    unsigned short* waT = (unsigned short*)ws; ws += (size_t)Cc * QKVW * 2;
    unsigned short* wpT = (unsigned short*)ws;

    // 0) one-shot bf16 casts (+ weight transposes to [N,K])
    cast_bf16<<<(M * Cc + 255) / 256, 256, 0, stream>>>(x, xb, M * Cc);
    cast_transpose_bf16<<<(Cc * QKVW + 255) / 256, 256, 0, stream>>>(
        w_attn, waT, Cc, QKVW);
    cast_transpose_bf16<<<(Cc * Cc + 255) / 256, 256, 0, stream>>>(
        w_proj, wpT, Cc, Cc);

    // 1) qkv = x @ w_attn
    gemm_bf16_wmma<<<dim3(M / 128, QKVW / 128), 256, 0, stream>>>(
        xb, waT, qkv, M, QKVW, Cc);

    // 2) RoPE + split + head-major transpose
    {
        const int n = Bc * Tc * (QKVW / 2);
        rope_split<<<(n + 255) / 256, 256, 0, stream>>>(qkv, f_cos, f_sin, Qb, Kb, Vb);
    }

    // 3) flash attention (bf16 out)
    attn_wmma<<<dim3(Tc / 64, Bc * NH), 128, 0, stream>>>(Qb, Kb, Vb, Obf);

    // 4) out = O @ w_proj
    gemm_bf16_wmma<<<dim3(M / 128, Cc / 128), 256, 0, stream>>>(
        Obf, wpT, out, M, Cc, Cc);
}
